// CIN_74921409511673
// MI455X (gfx1250) — matrix operand order repaired and checked by reading, not compiled, not run on hardware
//
#include <hip/hip_runtime.h>
#include <hip/hip_bf16.h>

typedef __attribute__((ext_vector_type(16))) _Float16 v16h;
typedef __attribute__((ext_vector_type(8)))  float    v8f;
typedef __attribute__((ext_vector_type(2)))  _Float16 h2;

#define NSAMP 2048
#define F0    64
#define EMBD  16

// ---------------------------------------------------------------------------
// prep_W: fp32 W [K x 128] -> f16 A-fragment layout for V_WMMA_F32_16X16X32_F16
// Wh layout: [mt(8)][ks(KS)][lane(32)][16 halves]
//   lane (<16): M=lane,  VGPR v holds K = {2v,2v+1} (v<8, i.e. K 0..15)
//   lane (>=16): M=lane-16, same + 16
//   A[m, k] = W[ks*32 + k, mt*16 + m]   (i.e. W transposed)
// ---------------------------------------------------------------------------
__global__ void prep_W(const float* __restrict__ W, _Float16* __restrict__ Wh,
                       int KS, int ntot) {
  int t = blockIdx.x * blockDim.x + threadIdx.x;
  if (t >= ntot) return;
  int v    = t & 7;
  int lane = (t >> 3) & 31;
  int rest = t >> 8;
  int ks   = rest % KS;
  int mt   = rest / KS;
  int col  = mt * 16 + (lane & 15);
  int kl   = ((lane >> 4) << 4) + (v << 1);
  int k    = ks * 32 + kl;
  h2 o;
  o.x = (_Float16)W[(size_t)k * 128 + col];
  o.y = (_Float16)W[(size_t)(k + 1) * 128 + col];
  *(h2*)(Wh + ((size_t)((mt * KS + ks) * 32 + lane)) * 16 + (v << 1)) = o;
}

// ---------------------------------------------------------------------------
// prep_X: fp32 x [B,64,16] -> f16 B-fragment layout (32x16 per fragment)
// Xh layout: [b][jf(2)][lane(32)][16 halves]
//   lane: N(col)=lane&15, K rows = (lane>>4)*16 + {2v,2v+1}
// ---------------------------------------------------------------------------
__global__ void prep_X(const float* __restrict__ x, _Float16* __restrict__ Xh,
                       int ntot) {
  int t = blockIdx.x * blockDim.x + threadIdx.x;
  if (t >= ntot) return;
  int v    = t & 7;
  int lane = (t >> 3) & 31;
  int jf   = (t >> 8) & 1;
  int b    = t >> 9;
  int d    = lane & 15;
  int j    = jf * 32 + ((lane >> 4) << 4) + (v << 1);
  const float* xb = x + (size_t)b * (F0 * EMBD);
  h2 o;
  o.x = (_Float16)xb[(size_t)j * EMBD + d];
  o.y = (_Float16)xb[(size_t)(j + 1) * EMBD + d];
  *(h2*)(Xh + ((size_t)((b * 2 + jf) * 32 + lane)) * 16 + (v << 1)) = o;
}

// ---------------------------------------------------------------------------
// One CIN layer for two samples (b0, b0+1) per wave.
//   C[128 x 16] = W^T @ P_b ; P_b[(i,j), d] = x[b,i,d] * xi[b,j,d]
// A fragment shared between the two samples (halves W bandwidth).
// ---------------------------------------------------------------------------
__device__ __forceinline__ void cin_layer_pair(
    const float*    __restrict__ x,    // x0 scalars [B,64,16] fp32
    const _Float16* __restrict__ Xi,   // B-frag source: nf frags / sample
    const _Float16* __restrict__ Wh,   // A frags [8][KS][32][16]
    _Float16*       __restrict__ Hout, // B-frag dest (nf_out=4) or nullptr
    float*          __restrict__ out,  // [B,384]
    int b0, int lane, int KS, int nfshift, int lay) {
  v8f zero = {};
  v8f C0[8], C1[8];
#pragma unroll
  for (int m = 0; m < 8; ++m) { C0[m] = zero; C1[m] = zero; }

  const int nf = 1 << nfshift;
  const int d  = lane & 15;
  const float*    x0a = x + (size_t)b0 * (F0 * EMBD) + d;
  const float*    x0b = x0a + (F0 * EMBD);
  const _Float16* xia = Xi + ((size_t)b0 * nf * 32 + lane) * 16;
  const _Float16* xib = xia + (size_t)nf * 32 * 16;

  for (int ks = 0; ks < KS; ++ks) {
    int i  = ks >> nfshift;        // shared x0 field for this 32-wide K step
    int jf = ks & (nf - 1);
    v16h f0 = *(const v16h*)(xia + (size_t)jf * 512);
    v16h f1 = *(const v16h*)(xib + (size_t)jf * 512);
    _Float16 s0 = (_Float16)x0a[i * EMBD];
    _Float16 s1 = (_Float16)x0b[i * EMBD];
    v16h B0 = f0 * s0;             // rank-1 build of P_b fragment
    v16h B1 = f1 * s1;
    const _Float16* wp = Wh + ((size_t)ks * 32 + lane) * 16;
#pragma unroll
    for (int mt = 0; mt < 8; ++mt) {
      v16h A = *(const v16h*)(wp + (size_t)mt * KS * 512);
      C0[mt] = __builtin_amdgcn_wmma_f32_16x16x32_f16(false, A, false, B0,
                                                      (short)0, C0[mt], false, false);
      C1[mt] = __builtin_amdgcn_wmma_f32_16x16x32_f16(false, A, false, B1,
                                                      (short)0, C1[mt], false, false);
    }
  }

  // Epilogue: store h (B-frag layout for next layer) + reduce over d -> out.
  int hi = lane >> 4;
#pragma unroll
  for (int mt = 0; mt < 8; ++mt) {
#pragma unroll
    for (int p = 0; p < 8; ++p) {
      int   r  = mt * 16 + p + (hi << 3);   // output channel row
      float v0 = C0[mt][p];
      float v1 = C1[mt][p];
      if (Hout) {
        int fj = r >> 5;
        int jl = r & 31;
        int lp = (jl & 16) | d;             // dest lane in B-frag
        int hv = jl & 15;                   // half index (2v+h)
        size_t off = ((size_t)fj * 32 + lp) * 16 + hv;
        Hout[(size_t)b0 * 2048 + off]       = (_Float16)v0;
        Hout[(size_t)(b0 + 1) * 2048 + off] = (_Float16)v1;
      }
      float t0 = v0, t1 = v1;
      t0 += __shfl_xor(t0, 1);  t1 += __shfl_xor(t1, 1);
      t0 += __shfl_xor(t0, 2);  t1 += __shfl_xor(t1, 2);
      t0 += __shfl_xor(t0, 4);  t1 += __shfl_xor(t1, 4);
      t0 += __shfl_xor(t0, 8);  t1 += __shfl_xor(t1, 8);
      if ((lane & 15) == 0) {
        out[(size_t)b0 * 384 + lay * 128 + r]       = t0;
        out[(size_t)(b0 + 1) * 384 + lay * 128 + r] = t1;
      }
    }
  }
}

__global__ __launch_bounds__(256) void cin_main(
    const float* __restrict__ x,
    const _Float16* __restrict__ Xh,
    const _Float16* __restrict__ Wh0,
    const _Float16* __restrict__ Wh1,
    const _Float16* __restrict__ Wh2,
    _Float16* __restrict__ H1,
    _Float16* __restrict__ H2,
    float* __restrict__ out) {
  int wid  = (blockIdx.x * blockDim.x + threadIdx.x) >> 5;
  int lane = threadIdx.x & 31;
  int b0   = wid * 2;
  cin_layer_pair(x, Xh, Wh0, H1, out, b0, lane, 128, 1, 0);
  asm volatile("s_wait_storecnt 0" ::: "memory");  // own h stores visible
  cin_layer_pair(x, H1, Wh1, H2, out, b0, lane, 256, 2, 1);
  asm volatile("s_wait_storecnt 0" ::: "memory");
  cin_layer_pair(x, H2, Wh2, nullptr, out, b0, lane, 256, 2, 2);
}

extern "C" void kernel_launch(void* const* d_in, const int* in_sizes, int n_in,
                              void* d_out, int out_size, void* d_ws, size_t ws_size,
                              hipStream_t stream) {
  const float* x  = (const float*)d_in[0];
  const float* W0 = (const float*)d_in[1];
  const float* W1 = (const float*)d_in[2];
  const float* W2 = (const float*)d_in[3];

  char* ws = (char*)d_ws;
  const size_t MB = 1u << 20;
  _Float16* Wh0 = (_Float16*)(ws + 0 * MB);   // 1 MB  (4096x128 f16)
  _Float16* Wh1 = (_Float16*)(ws + 1 * MB);   // 2 MB  (8192x128 f16)
  _Float16* Wh2 = (_Float16*)(ws + 3 * MB);   // 2 MB
  _Float16* Xh  = (_Float16*)(ws + 5 * MB);   // 4 MB  (x in B-frag f16)
  _Float16* H1  = (_Float16*)(ws + 9 * MB);   // 8 MB
  _Float16* H2  = (_Float16*)(ws + 17 * MB);  // 8 MB

  { int n = 8 * 128 * 32 * 8; prep_W<<<(n + 255) / 256, 256, 0, stream>>>(W0, Wh0, 128, n); }
  { int n = 8 * 256 * 32 * 8; prep_W<<<(n + 255) / 256, 256, 0, stream>>>(W1, Wh1, 256, n); }
  { int n = 8 * 256 * 32 * 8; prep_W<<<(n + 255) / 256, 256, 0, stream>>>(W2, Wh2, 256, n); }
  { int n = NSAMP * 2 * 32 * 8; prep_X<<<(n + 255) / 256, 256, 0, stream>>>(x, Xh, n); }

  // 2048 samples, 2 per wave -> 1024 waves -> 128 blocks x 256 threads
  cin_main<<<128, 256, 0, stream>>>(x, Xh, Wh0, Wh1, Wh2, H1, H2, (float*)d_out);
}
